// SelfAttention_35124242546827
// MI455X (gfx1250) — compile-verified, matrix operand order checked by reading, not run on hardware
//
#include <hip/hip_runtime.h>

typedef __bf16 bf16_t;
typedef __attribute__((ext_vector_type(16))) __bf16 v16bf;
typedef __attribute__((ext_vector_type(8)))  __bf16 v8bf;
typedef __attribute__((ext_vector_type(8)))  float  v8f;
typedef __attribute__((ext_vector_type(4)))  float  v4f;

#define WMMA_BF16(A, B, C) \
  __builtin_amdgcn_wmma_f32_16x16x32_bf16(false, (A), false, (B), (short)0, (C), false, false)

// A-fragment (16x32 bf16) / B-fragment (load B^T rows; identical layout, N in the lane role).
// Lane (r = lane&15, h = lane>>4): row r, K = {k0+h*8 .. +7} then {k0+16+h*8 .. +7}.
static __device__ __forceinline__ v16bf load_frag_bf16(const bf16_t* __restrict__ p,
                                                       int ld, int row, int k0, int lane) {
  const int r = lane & 15, h = lane >> 4;
  const bf16_t* base = p + (size_t)(row + r) * (size_t)ld + (size_t)(k0 + h * 8);
  v16bf a;
  a.lo = *(const v8bf*)(base);
  a.hi = *(const v8bf*)(base + 16);
  return a;
}

// Register-blocked WMMA GEMM core: one wave computes a (16*MF) x (16*NF) tile.
template <int MF, int NF>
static __device__ __forceinline__ void gemm_core(const bf16_t* __restrict__ A,
                                                 const bf16_t* __restrict__ Bt,
                                                 int lda, int ldb, int K,
                                                 int mt, int nt, int lane,
                                                 v8f (&C)[MF][NF]) {
  for (int k0 = 0; k0 < K; k0 += 32) {
    v16bf Af[MF];
#pragma unroll
    for (int mi = 0; mi < MF; ++mi)
      Af[mi] = load_frag_bf16(A, lda, mt + 16 * mi, k0, lane);
#pragma unroll
    for (int j = 0; j < NF; ++j) {
      v16bf Bf = load_frag_bf16(Bt, ldb, nt + 16 * j, k0, lane);
#pragma unroll
      for (int mi = 0; mi < MF; ++mi)
        C[mi][j] = WMMA_BF16(Af[mi], Bf, C[mi][j]);
    }
  }
}

// Elementwise fp32 -> bf16 (8 elements/thread, grid-stride).
__global__ __launch_bounds__(256) void cvt_f32_bf16_kernel(const float* __restrict__ in,
                                                           bf16_t* __restrict__ outp,
                                                           long long n) {
  long long idx = ((long long)blockIdx.x * 256 + threadIdx.x) * 8;
  const long long stride = (long long)gridDim.x * 256 * 8;
  for (; idx < n; idx += stride) {
    v8f f;
    f.lo = *(const v4f*)(in + idx);
    f.hi = *(const v4f*)(in + idx + 4);
    *(v8bf*)(outp + idx) = __builtin_convertvector(f, v8bf);
  }
}

// out = X * W^T + bias ; X:(M,K) bf16, W:(N,K) bf16 row-major.
// transposed==0: out (M,N) bf16 row-major; transposed==1: out (N,M) bf16 row-major (V^T).
// Wave tile 64x64 (4x4 fragments).
__global__ __launch_bounds__(256) void qkv_proj_kernel(
    const bf16_t* __restrict__ X, const bf16_t* __restrict__ W,
    const float* __restrict__ bias, bf16_t* __restrict__ out,
    int M, int N, int K, int transposed) {
  const int lane = threadIdx.x & 31;
  const int wid  = blockIdx.x * 8 + (threadIdx.x >> 5);
  const int tilesN = N >> 6;
  const int mt = (wid / tilesN) << 6;
  const int nt = (wid % tilesN) << 6;

  v8f C[4][4] = {};
  gemm_core<4, 4>(X, W, K, K, K, mt, nt, lane, C);

  const int r = lane & 15, h = lane >> 4;
#pragma unroll
  for (int mi = 0; mi < 4; ++mi) {
    const int rowb = mt + 16 * mi + h * 8;
#pragma unroll
    for (int j = 0; j < 4; ++j) {
      const int n = nt + 16 * j + r;
      const float bval = bias[n];
      if (!transposed) {
#pragma unroll
        for (int i = 0; i < 8; ++i)
          out[(size_t)(rowb + i) * (size_t)N + n] = (bf16_t)(C[mi][j][i] + bval);
      } else {
        v8bf o;
#pragma unroll
        for (int i = 0; i < 8; ++i) o[i] = (bf16_t)(C[mi][j][i] + bval);
        *(v8bf*)(out + (size_t)n * (size_t)M + (size_t)rowb) = o;
      }
    }
  }
}

// out = scale * (A * Bt^T), batched via blockIdx.y.
// A:(M,K) bf16 lda, Bt:(N,K) bf16 ldb, out:(M,N) f32 ldo. Wave tile 32x64 (2x4 fragments).
__global__ __launch_bounds__(256) void attn_gemm_kernel(
    const bf16_t* __restrict__ A, const bf16_t* __restrict__ Bt,
    float* __restrict__ outp, int M, int N, int K,
    int lda, int ldb, int ldo,
    long long strideA, long long strideB, long long strideO, float scale) {
  const int b = blockIdx.y;
  const bf16_t* Ab  = A  + (size_t)b * strideA;
  const bf16_t* Btb = Bt + (size_t)b * strideB;
  float* ob = outp + (size_t)b * strideO;

  const int lane = threadIdx.x & 31;
  const int wid  = blockIdx.x * 8 + (threadIdx.x >> 5);
  const int tilesN = N >> 6;
  const int mt = (wid / tilesN) << 5;
  const int nt = (wid % tilesN) << 6;

  v8f C[2][4] = {};
  gemm_core<2, 4>(Ab, Btb, lda, ldb, K, mt, nt, lane, C);

  const int r = lane & 15, h = lane >> 4;
#pragma unroll
  for (int mi = 0; mi < 2; ++mi) {
    const int rowb = mt + 16 * mi + h * 8;
#pragma unroll
    for (int j = 0; j < 4; ++j) {
      const int n = nt + 16 * j + r;
#pragma unroll
      for (int i = 0; i < 8; ++i)
        ob[(size_t)(rowb + i) * (size_t)ldo + n] = C[mi][j][i] * scale;
    }
  }
}

// Row softmax: fp32 scores (N==2048) -> bf16 probabilities. One 256-thread block per row,
// batched via blockIdx.y.
__global__ __launch_bounds__(256) void softmax_kernel(
    const float* __restrict__ S, bf16_t* __restrict__ P, int N,
    long long strideS, long long strideP) {
  const int row = blockIdx.x;
  const int tid = threadIdx.x;
  const float* srow = S + (size_t)blockIdx.y * strideS + (size_t)row * (size_t)N;
  bf16_t* prow = P + (size_t)blockIdx.y * strideP + (size_t)row * (size_t)N;
  __shared__ float red[8];

  float vals[8];
  float mx = -3.0e38f;
#pragma unroll
  for (int i = 0; i < 8; ++i) {
    vals[i] = srow[tid + (i << 8)];
    mx = fmaxf(mx, vals[i]);
  }
#pragma unroll
  for (int off = 16; off >= 1; off >>= 1)
    mx = fmaxf(mx, __shfl_xor(mx, off, 32));
  if ((tid & 31) == 0) red[tid >> 5] = mx;
  __syncthreads();
  mx = red[0];
#pragma unroll
  for (int w = 1; w < 8; ++w) mx = fmaxf(mx, red[w]);
  __syncthreads();

  float sum = 0.0f;
#pragma unroll
  for (int i = 0; i < 8; ++i) {
    vals[i] = __expf(vals[i] - mx);
    sum += vals[i];
  }
#pragma unroll
  for (int off = 16; off >= 1; off >>= 1)
    sum += __shfl_xor(sum, off, 32);
  if ((tid & 31) == 0) red[tid >> 5] = sum;
  __syncthreads();
  sum = 0.0f;
#pragma unroll
  for (int w = 0; w < 8; ++w) sum += red[w];
  const float inv = 1.0f / sum;

#pragma unroll
  for (int i = 0; i < 8; ++i)
    prow[tid + (i << 8)] = (bf16_t)(vals[i] * inv);
}

extern "C" void kernel_launch(void* const* d_in, const int* in_sizes, int n_in,
                              void* d_out, int out_size, void* d_ws, size_t ws_size,
                              hipStream_t stream) {
  const float* x  = (const float*)d_in[0];
  const float* Wq = (const float*)d_in[1];
  const float* bq = (const float*)d_in[2];
  const float* Wk = (const float*)d_in[3];
  const float* bk = (const float*)d_in[4];
  const float* Wv = (const float*)d_in[5];
  const float* bv = (const float*)d_in[6];
  float* out = (float*)d_out;

  const int B = 8, S = 2048, D = 1024;
  const int MS = B * S; // 16384
  const float scale = 0.03125f; // 1/sqrt(1024)

  // ---- workspace layout (all offsets 16B aligned) ----
  char* ws = (char*)d_ws;
  bf16_t* xb  = (bf16_t*)(ws);                         // 16384x1024 bf16 = 32 MB
  bf16_t* Wqb = (bf16_t*)(ws + (size_t)33554432);      // 1024x1024 bf16 = 2 MB
  bf16_t* Wkb = (bf16_t*)(ws + (size_t)35651584);
  bf16_t* Wvb = (bf16_t*)(ws + (size_t)37748736);
  bf16_t* Q   = (bf16_t*)(ws + (size_t)39845888);      // 32 MB
  bf16_t* Km  = (bf16_t*)(ws + (size_t)73400320);      // 32 MB
  bf16_t* VT  = (bf16_t*)(ws + (size_t)106954752);     // V^T (1024 x 16384), 32 MB
  const size_t attBase = 140509184;
  // Full tier: scores f32 for all batches (134 MB) + P bf16 for all batches (67 MB)
  const size_t fullNeed = attBase + (size_t)B * S * S * 4 + (size_t)B * S * S * 2; // ~326 MB

  dim3 blk(256);

  // ---- 1) convert inputs to bf16 ----
  cvt_f32_bf16_kernel<<<4096, blk, 0, stream>>>(x, xb, (long long)MS * D);
  cvt_f32_bf16_kernel<<<512,  blk, 0, stream>>>(Wq, Wqb, (long long)D * D);
  cvt_f32_bf16_kernel<<<512,  blk, 0, stream>>>(Wk, Wkb, (long long)D * D);
  cvt_f32_bf16_kernel<<<512,  blk, 0, stream>>>(Wv, Wvb, (long long)D * D);

  // ---- 2) projections (pure bf16 WMMA GEMM, 64x64 wave tiles) ----
  const int projBlocks = (MS / 64) * (D / 64) / 8; // 512
  qkv_proj_kernel<<<projBlocks, blk, 0, stream>>>(xb, Wqb, bq, Q,  MS, D, D, 0);
  qkv_proj_kernel<<<projBlocks, blk, 0, stream>>>(xb, Wkb, bk, Km, MS, D, D, 0);
  qkv_proj_kernel<<<projBlocks, blk, 0, stream>>>(xb, Wvb, bv, VT, MS, D, D, 1);

  const int scBlocks = (S / 32) * (S / 64) / 8; // 256 per batch
  const int pvBlocks = (S / 32) * (D / 64) / 8; // 128 per batch

  if (ws_size >= fullNeed) {
    // ---- full-parallel tier: all batches per launch (gridDim.y = 8) ----
    float*  Sc = (float*) (ws + attBase);
    bf16_t* P  = (bf16_t*)(ws + attBase + (size_t)B * S * S * 4);
    attn_gemm_kernel<<<dim3(scBlocks, B), blk, 0, stream>>>(
        Q, Km, Sc, S, S, D, D, D, S,
        (long long)S * D, (long long)S * D, (long long)S * S, scale);
    softmax_kernel<<<dim3(S, B), blk, 0, stream>>>(
        Sc, P, S, (long long)S * S, (long long)S * S);
    attn_gemm_kernel<<<dim3(pvBlocks, B), blk, 0, stream>>>(
        P, VT, out, S, D, S, S, MS, D,
        (long long)S * S, (long long)S, (long long)S * D, 1.0f);
  } else {
    // ---- per-batch tier: reuse one score/P buffer, stream-ordered ----
    float*  Sc = (float*) (ws + attBase);                        // 16 MB
    bf16_t* P  = (bf16_t*)(ws + attBase + (size_t)S * S * 4);    // 8 MB
    for (int b = 0; b < B; ++b) {
      const bf16_t* Qb = Q  + (size_t)b * S * D;
      const bf16_t* Kb = Km + (size_t)b * S * D;
      attn_gemm_kernel<<<dim3(scBlocks, 1), blk, 0, stream>>>(
          Qb, Kb, Sc, S, S, D, D, D, S, 0, 0, 0, scale);
      softmax_kernel<<<dim3(S, 1), blk, 0, stream>>>(Sc, P, S, 0, 0);
      attn_gemm_kernel<<<dim3(pvBlocks, 1), blk, 0, stream>>>(
          P, VT + (size_t)b * S, out + (size_t)b * S * D,
          S, D, S, S, MS, D, 0, 0, 0, 1.0f);
    }
  }
}
